// EmbedMatcher_28226525070102
// MI455X (gfx1250) — compile-verified, matrix operand order checked by reading, not run on hardware
//
#include <hip/hip_runtime.h>
#include <hip/hip_bf16.h>

typedef __attribute__((ext_vector_type(16))) _Float16 v16h;
typedef __attribute__((ext_vector_type(8)))  _Float16 v8h;
typedef __attribute__((ext_vector_type(8)))  float    v8f;

#define PAD_IDX 200000
#define KNB 10
#define MAXK 64
#define NQ 4096
#define NSUP 5

__device__ __forceinline__ float sigm(float x) { return 1.0f / (1.0f + __expf(-x)); }

// Load a 16-half WMMA fragment for one lane from a row-major row pointer.
// Per CDNA5 ISA 16-bit A/B layout: lanes 0-15 hold K[0:8)+K[16:24), lanes 16-31 hold K[8:16)+K[24:32).
__device__ __forceinline__ v16h load_frag16(const _Float16* p, int lhi) {
    v8h lo = *(const v8h*)(p + lhi * 8);
    v8h hi = *(const v8h*)(p + 16 + lhi * 8);
    v16h r;
#pragma unroll
    for (int i = 0; i < 8; ++i) { r[i] = lo[i]; r[i + 8] = hi[i]; }
    return r;
}

__device__ __forceinline__ v8f wmma16(v16h a, v16h b, v8f c) {
    return __builtin_amdgcn_wmma_f32_16x16x32_f16(false, a, false, b, (short)0, c, false, false);
}

// ---------------------------------------------------------------- weight convert
__global__ void convert_f16_kernel(const float* __restrict__ src, _Float16* __restrict__ dst, int n) {
    int i = blockIdx.x * 256 + threadIdx.x;
    if (i < n) dst[i] = (_Float16)src[i];
}

// ---------------------------------------------------------------- neighbor encoder
__global__ __launch_bounds__(256) void neighbor_kernel(
    const int* __restrict__ query, const int* __restrict__ support,
    const int* __restrict__ q_l1, const int* __restrict__ q_deg_l,
    const int* __restrict__ q_r1, const int* __restrict__ q_deg_r,
    const int* __restrict__ s_l1, const int* __restrict__ s_deg_l,
    const int* __restrict__ s_r1, const int* __restrict__ s_deg_r,
    const float* __restrict__ emb, const _Float16* __restrict__ gcnWh,
    const float* __restrict__ gcn_w_b, const float* __restrict__ gcn_b,
    const float* __restrict__ g1_w, const float* __restrict__ g1_b,
    const float* __restrict__ ln1_g, const float* __restrict__ ln1_b,
    const float* __restrict__ g2_w, const float* __restrict__ g2_b,
    const float* __restrict__ gate_temp,
    float* __restrict__ qvec, float* __restrict__ svec)
{
    __shared__ __align__(16) float ent[64][128];
    __shared__ int   conn_s[128];
    __shared__ float sim_s[64];
    __shared__ float selfv[128];
    __shared__ float red_s;
    __shared__ __align__(32) _Float16 Atile[16][256];
    __shared__ float agg_s[128];
    __shared__ int   sel_s[16];
    __shared__ int   cnt_s;
    __shared__ float pre_s[64];
    __shared__ float gate_s;

    const int bid = blockIdx.x, tid = threadIdx.x;
    const int* conn; int selfid, deg; float* outp;
    if (bid < NQ) {
        int r = bid; conn = q_l1 + r * MAXK * 2; deg = q_deg_l[r];
        selfid = query[r * 2 + 0]; outp = qvec + (size_t)r * 256;
    } else if (bid < 2 * NQ) {
        int r = bid - NQ; conn = q_r1 + r * MAXK * 2; deg = q_deg_r[r];
        selfid = query[r * 2 + 1]; outp = qvec + (size_t)r * 256 + 128;
    } else if (bid < 2 * NQ + NSUP) {
        int r = bid - 2 * NQ; conn = s_l1 + r * MAXK * 2; deg = s_deg_l[r];
        selfid = support[r * 2 + 0]; outp = svec + (size_t)r * 256;
    } else {
        int r = bid - 2 * NQ - NSUP; conn = s_r1 + r * MAXK * 2; deg = s_deg_r[r];
        selfid = support[r * 2 + 1]; outp = svec + (size_t)r * 256 + 128;
    }

    if (tid < 128) conn_s[tid] = conn[tid];
    else           selfv[tid - 128] = emb[(size_t)selfid * 128 + (tid - 128)];
    if (tid == 0) cnt_s = 0;
    __syncthreads();

    // gather 64 neighbor entity embeddings into LDS (float4)
    const float4* emb4 = (const float4*)emb;
    for (int i = tid; i < 64 * 32; i += 256) {
        int j = i >> 5, c4 = i & 31;
        int eid = conn_s[2 * j + 1];
        ((float4*)ent[j])[c4] = emb4[(size_t)eid * 32 + c4];
    }
    __syncthreads();

    const int lane = tid & 31, wv = tid >> 5;
    if (wv == 0) {
        float ss = 0.f;
        for (int c = lane; c < 128; c += 32) { float v = selfv[c]; ss += v * v; }
#pragma unroll
        for (int o = 16; o >= 1; o >>= 1) ss += __shfl_xor(ss, o, 32);
        if (lane == 0) red_s = ss;
    }
    __syncthreads();
    const float snorm = fmaxf(sqrtf(red_s), 1e-12f);

    for (int j = wv; j < 64; j += 8) {
        float d = 0.f, ee = 0.f;
        for (int c = lane; c < 128; c += 32) { float e = ent[j][c]; d += selfv[c] * e; ee += e * e; }
#pragma unroll
        for (int o = 16; o >= 1; o >>= 1) { d += __shfl_xor(d, o, 32); ee += __shfl_xor(ee, o, 32); }
        if (lane == 0) {
            float en = fmaxf(sqrtf(ee), 1e-12f);
            float s = d / (snorm * en);
            if (conn_s[2 * j] == PAD_IDX) s -= 1e9f;
            sim_s[j] = s;
        }
    }
    __syncthreads();

    // rank (matches top_k's tie-break by index); compact selected non-pad rows
    if (tid < 64) {
        float sj = sim_s[tid];
        int rank = 0;
#pragma unroll 8
        for (int i = 0; i < 64; ++i) {
            float si = sim_s[i];
            rank += (si > sj) || (si == sj && i < tid);
        }
        if (rank < KNB && conn_s[2 * tid] != PAD_IDX) {
            int slot = atomicAdd(&cnt_s, 1);
            sel_s[slot] = tid;
        }
    }
    __syncthreads();
    const int cnt = cnt_s;
    const float denom = fmaxf((float)cnt, 1.0f);

    // build compacted 16x256 f16 A tile: [rel_emb | ent_emb] for selected rows
    for (int i = tid; i < 16 * 256; i += 256) {
        int r = i >> 8, c = i & 255;
        float v = 0.f;
        if (r < cnt) {
            int j = sel_s[r];
            v = (c < 128) ? emb[(size_t)conn_s[2 * j] * 128 + c] : ent[j][c - 128];
        }
        Atile[r][c] = (_Float16)v;
    }
    __syncthreads();

    // GEMM 16x256 x (W^T: 256x128) via WMMA; wave w handles N tile w (cols w*16..)
    {
        const int lrow = lane & 15, lhi = lane >> 4;
        v8f acc = {};
        const _Float16* Brow = gcnWh + (size_t)(wv * 16 + lrow) * 256;
        const _Float16* Arow = &Atile[lrow][0];
#pragma unroll
        for (int kk = 0; kk < 8; ++kk) {
            v16h a = load_frag16(Arow + kk * 32, lhi);
            v16h b = load_frag16(Brow + kk * 32, lhi);
            acc = wmma16(a, b, acc);
        }
        int n = wv * 16 + lrow;
        float bias = gcn_w_b[n] + gcn_b[n];
        float part = 0.f;
#pragma unroll
        for (int i = 0; i < 8; ++i) {
            int m = i + 8 * lhi;
            if (m < cnt) {
                float v = acc[i] + bias;
                v = (v > 0.f) ? v : 0.01f * v;  // leaky_relu
                part += v;
            }
        }
        part += __shfl_xor(part, 16, 32);
        if (lhi == 0) agg_s[n] = part / denom;
    }
    __syncthreads();

    // gate MLP: 128 -> 64 (LN+relu) -> 1
    if (tid < 64) {
        float s = g1_b[tid];
        const float* wrow = g1_w + tid * 128;
#pragma unroll 4
        for (int c = 0; c < 128; ++c) s += wrow[c] * agg_s[c];
        pre_s[tid] = s;
    }
    __syncthreads();
    if (wv == 0) {
        float x1 = pre_s[lane], x2 = pre_s[lane + 32];
        float s = x1 + x2, sq = x1 * x1 + x2 * x2;
#pragma unroll
        for (int o = 16; o >= 1; o >>= 1) { s += __shfl_xor(s, o, 32); sq += __shfl_xor(sq, o, 32); }
        float m = s * (1.f / 64.f), var = sq * (1.f / 64.f) - m * m;
        float inv = rsqrtf(var + 1e-5f);
        float h1 = fmaxf((x1 - m) * inv * ln1_g[lane] + ln1_b[lane], 0.f);
        float h2 = fmaxf((x2 - m) * inv * ln1_g[lane + 32] + ln1_b[lane + 32], 0.f);
        float l = h1 * g2_w[lane] + h2 * g2_w[lane + 32];
#pragma unroll
        for (int o = 16; o >= 1; o >>= 1) l += __shfl_xor(l, o, 32);
        if (lane == 0) {
            l += g2_b[0];
            float t = fminf(fmaxf(gate_temp[0], 0.1f), 5.0f);
            gate_s = sigm(l / t) * ((deg > 0) ? 1.f : 0.f);
        }
    }
    __syncthreads();
    const float gate = gate_s;
    for (int c = tid; c < 128; c += 256)
        outp[c] = tanhf(selfv[c] + gate * agg_s[c]);
}

// ---------------------------------------------------------------- tiny support encoder (5 rows) + mean
__global__ __launch_bounds__(256) void support_enc_small_kernel(
    const float* __restrict__ svec,
    const float* __restrict__ p1w, const float* __restrict__ p1b,
    const float* __restrict__ p2w, const float* __restrict__ p2b,
    const float* __restrict__ lng, const float* __restrict__ lnb,
    float* __restrict__ sg)
{
    __shared__ float X[5][256];
    __shared__ float H[5][512];
    __shared__ float Y[5][256];
    __shared__ float Z[5][256];
    const int tid = threadIdx.x, lane = tid & 31, wv = tid >> 5;

    for (int i = tid; i < 5 * 256; i += 256) X[i >> 8][i & 255] = svec[i];
    __syncthreads();
    for (int i = tid; i < 5 * 512; i += 256) {
        int r = i >> 9, o = i & 511;
        float s = p1b[o];
        const float* wr = p1w + o * 256;
        for (int c = 0; c < 256; ++c) s += wr[c] * X[r][c];
        H[r][o] = fmaxf(s, 0.f);
    }
    __syncthreads();
    for (int i = tid; i < 5 * 256; i += 256) {
        int r = i >> 8, o = i & 255;
        float s = p2b[o];
        const float* wr = p2w + o * 512;
        for (int c = 0; c < 512; ++c) s += wr[c] * H[r][c];
        Y[r][o] = s + X[r][o];
    }
    __syncthreads();
    if (wv < 5) {
        int r = wv;
        float s = 0.f, sq = 0.f;
        for (int c = lane; c < 256; c += 32) { float v = Y[r][c]; s += v; sq += v * v; }
#pragma unroll
        for (int o = 16; o >= 1; o >>= 1) { s += __shfl_xor(s, o, 32); sq += __shfl_xor(sq, o, 32); }
        float m = s * (1.f / 256.f), var = sq * (1.f / 256.f) - m * m;
        float inv = rsqrtf(var + 1e-5f);
        for (int c = lane; c < 256; c += 32)
            Z[r][c] = (Y[r][c] - m) * inv * lng[c] + lnb[c];
    }
    __syncthreads();
    if (tid < 256) {
        float s = 0.f;
        for (int r = 0; r < 5; ++r) s += Z[r][tid];
        sg[tid] = s * 0.2f;
    }
}

// ---------------------------------------------------------------- batched support encoder for queries (WMMA)
__global__ __launch_bounds__(256) void support_enc_batch_kernel(
    const float* __restrict__ qvec,
    const _Float16* __restrict__ p1h, const float* __restrict__ p1b,
    const _Float16* __restrict__ p2h, const float* __restrict__ p2b,
    const float* __restrict__ lng, const float* __restrict__ lnb,
    float* __restrict__ qe32, _Float16* __restrict__ qe16)
{
    __shared__ __align__(32) _Float16 A0[32][256];   // 16 KB
    __shared__ __align__(32) _Float16 H1[32][512];   // 32 KB
    __shared__ float Y[32][256];                     // 32 KB
    const int tid = threadIdx.x, lane = tid & 31, wv = tid >> 5;
    const int lrow = lane & 15, lhi = lane >> 4;
    const int r0 = blockIdx.x * 32;

    for (int i = tid; i < 32 * 256; i += 256)
        A0[i >> 8][i & 255] = (_Float16)qvec[(size_t)(r0 + (i >> 8)) * 256 + (i & 255)];
    __syncthreads();

    // phase 1: h = relu(x @ p1^T + b1), 2 Mt x 32 Nt tiles, K=256
    for (int t = 0; t < 8; ++t) {
        int tile = wv * 8 + t, mt = tile & 1, nt = tile >> 1;
        v8f acc = {};
        const _Float16* Brow = p1h + (size_t)(nt * 16 + lrow) * 256;
        const _Float16* Arow = &A0[mt * 16 + lrow][0];
#pragma unroll
        for (int kk = 0; kk < 8; ++kk)
            acc = wmma16(load_frag16(Arow + kk * 32, lhi), load_frag16(Brow + kk * 32, lhi), acc);
        int n = nt * 16 + lrow;
        float bias = p1b[n];
#pragma unroll
        for (int i = 0; i < 8; ++i)
            H1[mt * 16 + i + 8 * lhi][n] = (_Float16)fmaxf(acc[i] + bias, 0.f);
    }
    __syncthreads();

    // phase 2: y = h @ p2^T + b2 + x, 2 Mt x 16 Nt tiles, K=512
    for (int t = 0; t < 4; ++t) {
        int tile = wv * 4 + t, mt = tile & 1, nt = tile >> 1;
        v8f acc = {};
        const _Float16* Brow = p2h + (size_t)(nt * 16 + lrow) * 512;
        const _Float16* Arow = &H1[mt * 16 + lrow][0];
#pragma unroll
        for (int kk = 0; kk < 16; ++kk)
            acc = wmma16(load_frag16(Arow + kk * 32, lhi), load_frag16(Brow + kk * 32, lhi), acc);
        int n = nt * 16 + lrow;
        float bias = p2b[n];
#pragma unroll
        for (int i = 0; i < 8; ++i) {
            int m = mt * 16 + i + 8 * lhi;
            Y[m][n] = acc[i] + bias + qvec[(size_t)(r0 + m) * 256 + n];
        }
    }
    __syncthreads();

    // layernorm per row, write f32 + f16
    for (int r = wv; r < 32; r += 8) {
        float s = 0.f, sq = 0.f;
        for (int c = lane; c < 256; c += 32) { float v = Y[r][c]; s += v; sq += v * v; }
#pragma unroll
        for (int o = 16; o >= 1; o >>= 1) { s += __shfl_xor(s, o, 32); sq += __shfl_xor(sq, o, 32); }
        float m = s * (1.f / 256.f), var = sq * (1.f / 256.f) - m * m;
        float inv = rsqrtf(var + 1e-5f);
        for (int c = lane; c < 256; c += 32) {
            float v = (Y[r][c] - m) * inv * lng[c] + lnb[c];
            qe32[(size_t)(r0 + r) * 256 + c] = v;
            qe16[(size_t)(r0 + r) * 256 + c] = (_Float16)v;
        }
    }
}

// ---------------------------------------------------------------- constant gate-bias vectors
__global__ void hr_const_kernel(const float* __restrict__ whh,
                                const float* __restrict__ b_ih, const float* __restrict__ b_hh,
                                const float* __restrict__ sg,
                                float* __restrict__ bias0, float* __restrict__ hrconst)
{
    int j = blockIdx.x * 256 + threadIdx.x;
    if (j < 2048) {
        float b0 = b_ih[j] + b_hh[j];
        const float* wr = whh + (size_t)j * 512 + 256;
        float s = 0.f;
        for (int c = 0; c < 256; ++c) s += wr[c] * sg[c];
        bias0[j] = b0;
        hrconst[j] = b0 + s;
    }
}

// ---------------------------------------------------------------- LSTM-like recurrence step (WMMA)
__global__ __launch_bounds__(256) void lstm_step_kernel(
    const _Float16* __restrict__ qe16, const float* __restrict__ qe32,
    const _Float16* __restrict__ h16_in,
    const _Float16* __restrict__ wihh, const _Float16* __restrict__ whhh,
    const float* __restrict__ bias0, const float* __restrict__ hrconst,
    float* __restrict__ c_st, _Float16* __restrict__ h16_out, float* __restrict__ h32_out,
    int step)
{
    __shared__ __align__(32) _Float16 A[32][512];  // [q_f16 | h_prev_f16], 32 KB
    __shared__ float G[4][32][64];                 // i,f,g,o tiles, 32 KB
    const int tid = threadIdx.x, lane = tid & 31, wv = tid >> 5;
    const int lrow = lane & 15, lhi = lane >> 4;
    const int rt = blockIdx.x >> 3, ct = blockIdx.x & 7;
    const int r0 = rt * 32, c0 = ct * 64;

    for (int i = tid; i < 32 * 256; i += 256) {
        int r = i >> 8, c = i & 255;
        A[r][c] = qe16[(size_t)(r0 + r) * 256 + c];
        A[r][c + 256] = (step == 0) ? (_Float16)0.f : h16_in[(size_t)(r0 + r) * 256 + c];
    }
    __syncthreads();

    const int gate = wv >> 1, nh = wv & 1;
    const float* bvec = (step == 0) ? bias0 : hrconst;
    for (int t = 0; t < 4; ++t) {
        int mt = t & 1, nt = nh * 2 + (t >> 1);
        v8f acc = {};
        int j = gate * 512 + c0 + nt * 16 + lrow;  // weight row = gate output column
        const _Float16* Bi = wihh + (size_t)j * 256;
        const _Float16* Bh = whhh + (size_t)j * 512;  // only first 256 cols used (h part)
        const _Float16* Arow = &A[mt * 16 + lrow][0];
#pragma unroll
        for (int kk = 0; kk < 8; ++kk)
            acc = wmma16(load_frag16(Arow + kk * 32, lhi), load_frag16(Bi + kk * 32, lhi), acc);
#pragma unroll
        for (int kk = 0; kk < 8; ++kk)
            acc = wmma16(load_frag16(Arow + 256 + kk * 32, lhi), load_frag16(Bh + kk * 32, lhi), acc);
        float bias = bvec[j];
#pragma unroll
        for (int i = 0; i < 8; ++i)
            G[gate][mt * 16 + i + 8 * lhi][nt * 16 + lrow] = acc[i] + bias;
    }
    __syncthreads();

    for (int i = tid; i < 32 * 64; i += 256) {
        int m = i >> 6, n = i & 63;
        int row = r0 + m, col = c0 + n;
        float iv = G[0][m][n], fv = G[1][m][n], gv = G[2][m][n], ov = G[3][m][n];
        float cold = (step == 0) ? 0.f : c_st[(size_t)row * 512 + col];
        float cn = sigm(fv) * cold + sigm(iv) * tanhf(gv);
        c_st[(size_t)row * 512 + col] = cn;
        float hv = sigm(ov) * tanhf(cn);
        if (col < 256) {
            float h = qe32[(size_t)row * 256 + col] + hv;
            h16_out[(size_t)row * 256 + col] = (_Float16)h;
            h32_out[(size_t)row * 256 + col] = h;
        }
    }
}

// ---------------------------------------------------------------- final dot
__global__ __launch_bounds__(256) void final_dot_kernel(const float* __restrict__ h32,
                                                        const float* __restrict__ sg,
                                                        float* __restrict__ out)
{
    const int tid = threadIdx.x, lane = tid & 31, wv = tid >> 5;
    int row = blockIdx.x * 8 + wv;
    float s = 0.f;
    for (int c = lane; c < 256; c += 32) s += h32[(size_t)row * 256 + c] * sg[c];
#pragma unroll
    for (int o = 16; o >= 1; o >>= 1) s += __shfl_xor(s, o, 32);
    if (lane == 0) out[row] = s;
}

// ---------------------------------------------------------------- launch
extern "C" void kernel_launch(void* const* d_in, const int* in_sizes, int n_in,
                              void* d_out, int out_size, void* d_ws, size_t ws_size,
                              hipStream_t stream) {
    const int*   query     = (const int*)d_in[0];
    const int*   support   = (const int*)d_in[1];
    const int*   q_l1      = (const int*)d_in[2];
    const int*   q_deg_l   = (const int*)d_in[3];
    const int*   q_r1      = (const int*)d_in[4];
    const int*   q_deg_r   = (const int*)d_in[5];
    const int*   s_l1      = (const int*)d_in[6];
    const int*   s_deg_l   = (const int*)d_in[7];
    const int*   s_r1      = (const int*)d_in[8];
    const int*   s_deg_r   = (const int*)d_in[9];
    const float* symbol_emb= (const float*)d_in[10];
    const float* gcn_w_w   = (const float*)d_in[11];
    const float* gcn_w_b   = (const float*)d_in[12];
    const float* gcn_b     = (const float*)d_in[13];
    const float* g1_w      = (const float*)d_in[14];
    const float* g1_b      = (const float*)d_in[15];
    const float* ln1_g     = (const float*)d_in[16];
    const float* ln1_b     = (const float*)d_in[17];
    const float* g2_w      = (const float*)d_in[18];
    const float* g2_b      = (const float*)d_in[19];
    const float* gate_temp = (const float*)d_in[20];
    const float* se_p1_w   = (const float*)d_in[21];
    const float* se_p1_b   = (const float*)d_in[22];
    const float* se_p2_w   = (const float*)d_in[23];
    const float* se_p2_b   = (const float*)d_in[24];
    const float* se_ln_g   = (const float*)d_in[25];
    const float* se_ln_b   = (const float*)d_in[26];
    const float* w_ih      = (const float*)d_in[27];
    const float* w_hh      = (const float*)d_in[28];
    const float* b_ih      = (const float*)d_in[29];
    const float* b_hh      = (const float*)d_in[30];

    char* ws = (char*)d_ws;
    size_t off = 0;
    auto take = [&](size_t bytes) -> void* {
        void* p = ws + off;
        off += (bytes + 255) & ~(size_t)255;
        return p;
    };
    _Float16* gcnWh   = (_Float16*)take(128 * 256 * sizeof(_Float16));
    _Float16* p1h     = (_Float16*)take(512 * 256 * sizeof(_Float16));
    _Float16* p2h     = (_Float16*)take(256 * 512 * sizeof(_Float16));
    _Float16* wihh    = (_Float16*)take(2048 * 256 * sizeof(_Float16));
    _Float16* whhh    = (_Float16*)take(2048 * 512 * sizeof(_Float16));
    float*    qvec    = (float*)take((size_t)NQ * 256 * sizeof(float));
    float*    svec    = (float*)take((size_t)NSUP * 256 * sizeof(float));
    float*    sg      = (float*)take(256 * sizeof(float));
    float*    qe32    = (float*)take((size_t)NQ * 256 * sizeof(float));
    _Float16* qe16    = (_Float16*)take((size_t)NQ * 256 * sizeof(_Float16));
    _Float16* hb0     = (_Float16*)take((size_t)NQ * 256 * sizeof(_Float16));
    _Float16* hb1     = (_Float16*)take((size_t)NQ * 256 * sizeof(_Float16));
    float*    h32     = (float*)take((size_t)NQ * 256 * sizeof(float));
    float*    c_st    = (float*)take((size_t)NQ * 512 * sizeof(float));
    float*    bias0   = (float*)take(2048 * sizeof(float));
    float*    hrconst = (float*)take(2048 * sizeof(float));
    (void)ws_size; (void)n_in; (void)in_sizes; (void)out_size;

    // 1) weight f32 -> f16 conversion
    convert_f16_kernel<<<(128 * 256 + 255) / 256, 256, 0, stream>>>(gcn_w_w, gcnWh, 128 * 256);
    convert_f16_kernel<<<(512 * 256 + 255) / 256, 256, 0, stream>>>(se_p1_w, p1h, 512 * 256);
    convert_f16_kernel<<<(256 * 512 + 255) / 256, 256, 0, stream>>>(se_p2_w, p2h, 256 * 512);
    convert_f16_kernel<<<(2048 * 256 + 255) / 256, 256, 0, stream>>>(w_ih, wihh, 2048 * 256);
    convert_f16_kernel<<<(2048 * 512 + 255) / 256, 256, 0, stream>>>(w_hh, whhh, 2048 * 512);

    // 2) neighbor encoders (q-left, q-right, s-left, s-right)
    neighbor_kernel<<<2 * NQ + 2 * NSUP, 256, 0, stream>>>(
        query, support, q_l1, q_deg_l, q_r1, q_deg_r, s_l1, s_deg_l, s_r1, s_deg_r,
        symbol_emb, gcnWh, gcn_w_b, gcn_b, g1_w, g1_b, ln1_g, ln1_b, g2_w, g2_b,
        gate_temp, qvec, svec);

    // 3) support encoder on the 5 support rows + mean -> sg[256]
    support_enc_small_kernel<<<1, 256, 0, stream>>>(svec, se_p1_w, se_p1_b, se_p2_w, se_p2_b,
                                                    se_ln_g, se_ln_b, sg);

    // 4) support encoder on queries (WMMA) -> qe32/qe16
    support_enc_batch_kernel<<<NQ / 32, 256, 0, stream>>>(qvec, p1h, se_p1_b, p2h, se_p2_b,
                                                          se_ln_g, se_ln_b, qe32, qe16);

    // 5) constant gate biases (attn collapses: r == support_g)
    hr_const_kernel<<<8, 256, 0, stream>>>(w_hh, b_ih, b_hh, sg, bias0, hrconst);

    // 6) 4 recurrence steps (double-buffered h in f16)
    _Float16* hbuf[2] = {hb0, hb1};
    for (int step = 0; step < 4; ++step) {
        const _Float16* hin = (step == 0) ? hb0 : hbuf[(step + 1) & 1];
        _Float16* hout = hbuf[step & 1];
        lstm_step_kernel<<<(NQ / 32) * 8, 256, 0, stream>>>(qe16, qe32, hin, wihh, whhh,
                                                            bias0, hrconst, c_st, hout, h32, step);
    }

    // 7) out[row] = dot(h_final[row], support_g)
    final_dot_kernel<<<NQ / 8, 256, 0, stream>>>(h32, sg, (float*)d_out);
}
